// CausalMultiHeadAttention_80453327389012
// MI455X (gfx1250) — matrix operand order repaired and checked by reading, not compiled, not run on hardware
//
#include <hip/hip_runtime.h>
#include <hip/hip_bf16.h>
#include <math.h>

// ---------------------------------------------------------------------------
// Causal MHA + RoPE for MI455X (gfx1250): wave32, V_WMMA_F32_16X16X32_F16,
// double-buffered GLOBAL_LOAD_ASYNC_TO_LDS_B128 staging (ASYNCcnt pipelined).
// ---------------------------------------------------------------------------

#define D_MODEL   2048
#define NUM_HEADS 16
#define D_K       128
#define SEQ       2048
#define BATCH     2
#define M_TOTAL   (BATCH * SEQ)
#define SCALE_QK  0.08838834764831845f   // 1/sqrt(128)
#define LN_THETA_OVER_DK 0.07195578373f  // ln(10000)/128

typedef __attribute__((ext_vector_type(16))) _Float16 v16h;
typedef __attribute__((ext_vector_type(8)))  _Float16 v8h;
typedef __attribute__((ext_vector_type(8)))  float    v8f;

#define WMMA_F16(a, b, c) \
    __builtin_amdgcn_wmma_f32_16x16x32_f16(false, (a), false, (b), (short)0, (c), false, false)

// ---- CDNA5 async LDS copy via inline asm (ASYNCcnt-tracked) ---------------
// GV mode: LDS address comes from the vdst VGPR, global address from the 64-bit vaddr pair.
// Flat shared pointers carry the LDS byte offset in addr[31:0] (aperture rule),
// so truncation yields the correct VDST value.
__device__ __forceinline__ void async_ld16(const void* g, void* l) {
    asm volatile("global_load_async_to_lds_b128 %0, %1, off"
                 :: "v"((unsigned int)(size_t)l),
                    "v"((unsigned long long)(size_t)g)
                 : "memory");
}

#define WAIT_ASYNC_8() asm volatile("s_wait_asynccnt 0x8" ::: "memory")
#define WAIT_ASYNC_0() asm volatile("s_wait_asynccnt 0x0" ::: "memory")

// ---------------------------------------------------------------------------
// 1) float32 -> float16 convert (for x)
// ---------------------------------------------------------------------------
__global__ void mha_cvt_f32_f16(const float* __restrict__ in,
                                _Float16* __restrict__ out, int n) {
    int i = blockIdx.x * blockDim.x + threadIdx.x;
    int stride = gridDim.x * blockDim.x;
    for (; i < n; i += stride) out[i] = (_Float16)in[i];
}

// ---------------------------------------------------------------------------
// 2) weight transpose + convert:  Wt[n][k] = (f16) W[k][n]   (2048x2048)
// ---------------------------------------------------------------------------
__global__ void mha_transpose_w(const float* __restrict__ W,
                                _Float16* __restrict__ Wt) {
    __shared__ float tile[32][33];
    int k0 = blockIdx.x * 32, n0 = blockIdx.y * 32;
    int tx = threadIdx.x & 31, ty = threadIdx.x >> 5;   // 32 x 8
    for (int i = 0; i < 32; i += 8)
        tile[ty + i][tx] = W[(size_t)(k0 + ty + i) * D_MODEL + n0 + tx];
    __syncthreads();
    for (int i = 0; i < 32; i += 8)
        Wt[(size_t)(n0 + ty + i) * D_MODEL + k0 + tx] = (_Float16)tile[tx][ty + i];
}

// ---------------------------------------------------------------------------
// 3) Tiled WMMA GEMM with double-buffered async LDS staging.
//    out[M x 2048] = A[M x 2048] * B, B given transposed as Bt[n][k] (f16).
//    mode 0: f16 plain; mode 1: f16 head-major [B][H][S][128]; mode 2: f32.
// ---------------------------------------------------------------------------
__global__ void __launch_bounds__(256)
mha_gemm128(const _Float16* __restrict__ A, const _Float16* __restrict__ Bt,
            void* __restrict__ outp, int mode) {
    __shared__ _Float16 lA[2][128][72];   // padded rows, conflict-free frags
    __shared__ _Float16 lB[2][128][72];

    const int m0 = blockIdx.x * 128, n0 = blockIdx.y * 128;
    const int t = threadIdx.x, lane = t & 31, wid = t >> 5;
    const int half = lane >> 4, l16 = lane & 15;
    const int wm = (wid & 3) * 32;       // 4 waves along M (32 rows each)
    const int wn = (wid >> 2) * 64;      // 2 waves along N (64 cols each)

    v8f acc[2][4];
    for (int mt = 0; mt < 2; mt++)
        for (int nt = 0; nt < 4; nt++)
            acc[mt][nt] = (v8f){0.f,0.f,0.f,0.f,0.f,0.f,0.f,0.f};

    auto stage = [&](int buf, int k0) {
        for (int i = 0; i < 4; i++) {            // 8 async b128 per thread
            int lin = i * 2048 + t * 8;
            int r = lin >> 6, c = lin & 63;
            async_ld16(A  + (size_t)(m0 + r) * D_MODEL + k0 + c, &lA[buf][r][c]);
            async_ld16(Bt + (size_t)(n0 + r) * D_MODEL + k0 + c, &lB[buf][r][c]);
        }
    };

    stage(0, 0);                                  // prologue prefetch
    const int NK = D_MODEL / 64;                  // 32 chunks
    for (int kc = 0; kc < NK; kc++) {
        const int buf = kc & 1;
        __syncthreads();                          // readers of buf^1 are done
        if (kc + 1 < NK) {
            stage(buf ^ 1, (kc + 1) * 64);        // prefetch next chunk
            WAIT_ASYNC_8();                       // current chunk landed
        } else {
            WAIT_ASYNC_0();
        }
        __syncthreads();                          // chunk kc visible block-wide

        for (int kk = 0; kk < 64; kk += 32) {
            v16h af[2];
            for (int mt = 0; mt < 2; mt++) {      // A frag: chunks {8h},{16+8h}
                const _Float16* p = &lA[buf][wm + mt * 16 + l16][kk + 8 * half];
                union { v16h v; v8h h2[2]; } u;
                u.h2[0] = *(const v8h*)p;
                u.h2[1] = *(const v8h*)(p + 16);
                af[mt] = u.v;
            }
            for (int nt = 0; nt < 4; nt++) {      // B frag: chunks {8h},{16+8h}
                const _Float16* p = &lB[buf][wn + nt * 16 + l16][kk + 8 * half];
                union { v16h v; v8h h2[2]; } u;
                u.h2[0] = *(const v8h*)p;
                u.h2[1] = *(const v8h*)(p + 16);
                for (int mt = 0; mt < 2; mt++)
                    acc[mt][nt] = WMMA_F16(af[mt], u.v, acc[mt][nt]);
            }
        }
    }

    // epilogue: C layout -> row = vg + 8*half, col = lane&15
    for (int mt = 0; mt < 2; mt++)
        for (int nt = 0; nt < 4; nt++)
            for (int vg = 0; vg < 8; vg++) {
                int m  = m0 + wm + mt * 16 + 8 * half + vg;
                int nl = wn + nt * 16 + l16;
                float val = acc[mt][nt][vg];
                if (mode == 0) {
                    ((_Float16*)outp)[(size_t)m * D_MODEL + n0 + nl] = (_Float16)val;
                } else if (mode == 1) {
                    int bb = m >> 11, s = m & (SEQ - 1), head = n0 >> 7;
                    ((_Float16*)outp)[(((size_t)(bb * NUM_HEADS + head)) * SEQ + s) * D_K + nl] =
                        (_Float16)val;
                } else {
                    ((float*)outp)[(size_t)m * D_MODEL + n0 + nl] = val;
                }
            }
}

// ---------------------------------------------------------------------------
// 4) RoPE: read plain f16 [B,S,2048], rotate pairs, write head-major f16
// ---------------------------------------------------------------------------
__global__ void mha_rope(const _Float16* __restrict__ in,
                         const int* __restrict__ pos,
                         _Float16* __restrict__ out) {
    int pid = blockIdx.x * blockDim.x + threadIdx.x;   // [0, B*S*1024)
    int b    = pid >> 21;
    int rem  = pid & ((1 << 21) - 1);
    int s    = rem >> 10;
    int hp   = rem & 1023;
    int head = hp >> 6;
    int j    = hp & 63;

    float freq = __expf(-(float)(2 * j) * LN_THETA_OVER_DK);
    float ang  = (float)pos[s] * freq;
    float sn, cs;
    __sincosf(ang, &sn, &cs);

    size_t ioff = ((size_t)(b * SEQ + s)) * D_MODEL + head * D_K + 2 * j;
    float xe = (float)in[ioff];
    float xo = (float)in[ioff + 1];

    size_t ooff = (((size_t)(b * NUM_HEADS + head)) * SEQ + s) * D_K + 2 * j;
    out[ooff]     = (_Float16)(xe * cs - xo * sn);
    out[ooff + 1] = (_Float16)(xo * cs + xe * sn);
}

// ---------------------------------------------------------------------------
// 5) Flash attention: one block per (b*h, 128 query rows). 8 waves, each owns
//    16 query rows. K tile staged via async LDS copy; V transposed manually.
// ---------------------------------------------------------------------------
__global__ void __launch_bounds__(256)
mha_attention(const _Float16* __restrict__ qh, const _Float16* __restrict__ kh,
              const _Float16* __restrict__ vh, _Float16* __restrict__ oh) {
    __shared__ _Float16 lK[128][136];       // [key][d]
    __shared__ _Float16 lV[128][136];       // transposed: [d][key]
    __shared__ _Float16 lP[8][16][136];     // per-wave P transpose scratch

    const int bh = blockIdx.y;
    const int b = bh >> 4, h = bh & 15;
    const int q0 = blockIdx.x * 128;
    const _Float16* Q = qh + (size_t)bh * SEQ * D_K;
    const _Float16* K = kh + (size_t)bh * SEQ * D_K;
    const _Float16* V = vh + (size_t)bh * SEQ * D_K;

    const int t = threadIdx.x, lane = t & 31, wid = t >> 5;
    const int half = lane >> 4, l16 = lane & 15;

    // Q fragments for this wave's 16 rows (4 chunks of K=32 along d)
    v16h qf[4];
    {
        int qr = q0 + wid * 16 + l16;
        for (int kf = 0; kf < 4; kf++) {
            const _Float16* p = Q + (size_t)qr * D_K + kf * 32 + 8 * half;
            union { v16h v; v8h h2[2]; } u;
            u.h2[0] = *(const v8h*)p;
            u.h2[1] = *(const v8h*)(p + 16);
            qf[kf] = u.v;
        }
    }

    v8f o[8];
    float m_i[8], l_i[8];
    for (int i = 0; i < 8; i++) {
        o[i] = (v8f){0.f,0.f,0.f,0.f,0.f,0.f,0.f,0.f};
        m_i[i] = -1e30f;
        l_i[i] = 0.f;
    }
    const int qrow_c = q0 + wid * 16 + 8 * half;   // + vg gives this lane's row

    for (int kb = 0; kb <= q0; kb += 128) {
        __syncthreads();
        // stage K tile [128 keys][128 d] via async LDS copy
        for (int i = 0; i < 8; i++) {
            int lin = i * 2048 + t * 8;
            int r = lin >> 7, c = lin & 127;
            async_ld16(K + (size_t)(kb + r) * D_K + c, &lK[r][c]);
        }
        // stage V transposed [128 d][128 keys] (manual: reshuffle needed)
        for (int i = 0; i < 8; i++) {
            int lin = i * 2048 + t * 8;
            int r = lin >> 7, c = lin & 127;
            v8h tmp = *(const v8h*)(V + (size_t)(kb + r) * D_K + c);
            for (int j = 0; j < 8; j++) lV[c + j][r] = tmp[j];
        }
        WAIT_ASYNC_0();
        __syncthreads();

        // S = Q * K^T  (contraction over d)
        v8f sa[8];
        for (int nt = 0; nt < 8; nt++) {
            v8f s = (v8f){0.f,0.f,0.f,0.f,0.f,0.f,0.f,0.f};
            for (int kf = 0; kf < 4; kf++) {
                const _Float16* p = &lK[nt * 16 + l16][kf * 32 + 8 * half];
                union { v16h v; v8h h2[2]; } u;
                u.h2[0] = *(const v8h*)p;
                u.h2[1] = *(const v8h*)(p + 16);
                s = WMMA_F16(qf[kf], u.v, s);
            }
            sa[nt] = s;
        }

        // online softmax per row (vg); 16-lane shfl reductions within half-wave
        for (int vg = 0; vg < 8; vg++) {
            int qr = qrow_c + vg;
            float pv[8];
            float mx = -1e30f;
            for (int nt = 0; nt < 8; nt++) {
                int key = kb + nt * 16 + l16;
                float sv = sa[nt][vg] * SCALE_QK;
                sv = (key <= qr) ? sv : -1e30f;
                pv[nt] = sv;
                mx = fmaxf(mx, sv);
            }
            for (int msk = 1; msk < 16; msk <<= 1)
                mx = fmaxf(mx, __shfl_xor(mx, msk, 32));
            float mnew  = fmaxf(m_i[vg], mx);
            float alpha = __expf(m_i[vg] - mnew);
            float lsum  = 0.f;
            for (int nt = 0; nt < 8; nt++) {
                float p = __expf(pv[nt] - mnew);
                lsum += p;
                lP[wid][vg + 8 * half][nt * 16 + l16] = (_Float16)p;
            }
            for (int msk = 1; msk < 16; msk <<= 1)
                lsum += __shfl_xor(lsum, msk, 32);
            l_i[vg] = l_i[vg] * alpha + lsum;
            m_i[vg] = mnew;
            for (int dt = 0; dt < 8; dt++)
                o[dt][vg] = o[dt][vg] * alpha;
        }

        // O += P * V  (contraction over keys; same-wave LDS is in-order)
        for (int kf = 0; kf < 4; kf++) {
            const _Float16* pa = &lP[wid][l16][kf * 32 + 8 * half];
            union { v16h v; v8h h2[2]; } ua;
            ua.h2[0] = *(const v8h*)pa;
            ua.h2[1] = *(const v8h*)(pa + 16);
            for (int dt = 0; dt < 8; dt++) {
                const _Float16* pb = &lV[dt * 16 + l16][kf * 32 + 8 * half];
                union { v16h v; v8h h2[2]; } ub;
                ub.h2[0] = *(const v8h*)pb;
                ub.h2[1] = *(const v8h*)(pb + 16);
                o[dt] = WMMA_F16(ua.v, ub.v, o[dt]);
            }
        }
    }

    // normalize + write head-gathered [B,S,2048] f16
    for (int dt = 0; dt < 8; dt++)
        for (int vg = 0; vg < 8; vg++) {
            int srow = qrow_c + vg;
            float val = o[dt][vg] / l_i[vg];
            oh[((size_t)(b * SEQ + srow)) * D_MODEL + h * D_K + dt * 16 + l16] =
                (_Float16)val;
        }
}

// ---------------------------------------------------------------------------
// host launcher
// ---------------------------------------------------------------------------
extern "C" void kernel_launch(void* const* d_in, const int* in_sizes, int n_in,
                              void* d_out, int out_size, void* d_ws, size_t ws_size,
                              hipStream_t stream) {
    const float* x  = (const float*)d_in[0];
    const int*   tp = (const int*)d_in[1];
    const float* Wq = (const float*)d_in[2];
    const float* Wk = (const float*)d_in[3];
    const float* Wv = (const float*)d_in[4];
    const float* Wo = (const float*)d_in[5];
    float* out = (float*)d_out;

    char* w = (char*)d_ws;
    const size_t XE  = (size_t)M_TOTAL * D_MODEL;
    const size_t WE  = (size_t)D_MODEL * D_MODEL;
    _Float16* xh   = (_Float16*)(w);
    _Float16* WqT  = (_Float16*)(w + XE * 2);
    _Float16* WkT  = WqT + WE;
    _Float16* WvT  = WkT + WE;
    _Float16* WoT  = WvT + WE;
    _Float16* qpre = WoT + WE;
    _Float16* kpre = qpre + XE;
    _Float16* qhm  = kpre + XE;
    _Float16* khm  = qhm + XE;
    _Float16* vhm  = khm + XE;
    _Float16* ohb  = qpre;                                   // reuse dead qpre

    mha_cvt_f32_f16<<<4096, 256, 0, stream>>>(x, xh, (int)XE);
    dim3 tg(64, 64);
    mha_transpose_w<<<tg, 256, 0, stream>>>(Wq, WqT);
    mha_transpose_w<<<tg, 256, 0, stream>>>(Wk, WkT);
    mha_transpose_w<<<tg, 256, 0, stream>>>(Wv, WvT);
    mha_transpose_w<<<tg, 256, 0, stream>>>(Wo, WoT);

    dim3 gg(M_TOTAL / 128, D_MODEL / 128);                   // (32,16)
    mha_gemm128<<<gg, 256, 0, stream>>>(xh, WqT, (void*)qpre, 0);
    mha_gemm128<<<gg, 256, 0, stream>>>(xh, WkT, (void*)kpre, 0);
    mha_gemm128<<<gg, 256, 0, stream>>>(xh, WvT, (void*)vhm, 1);

    const int pairs = (int)(XE / 2);
    mha_rope<<<pairs / 256, 256, 0, stream>>>(qpre, tp, qhm);
    mha_rope<<<pairs / 256, 256, 0, stream>>>(kpre, tp, khm);

    dim3 ga(SEQ / 128, BATCH * NUM_HEADS);                   // (16,32)
    mha_attention<<<ga, 256, 0, stream>>>(qhm, khm, vhm, ohb);

    mha_gemm128<<<gg, 256, 0, stream>>>(ohb, WoT, (void*)out, 2);
}
